// down_block_46428596470052
// MI455X (gfx1250) — matrix operand order repaired and checked by reading, not compiled, not run on hardware
//
#include <hip/hip_runtime.h>
#include <hip/hip_bf16.h>

// ---------------------------------------------------------------------------
// CDNA5 (gfx1250) implementation of the SS-Conv-SSM down block.
// - All channel contractions (patch-merge reduction, 3x3x3 convs as implicit
//   GEMM, in_proj / x_proj / cC / out_proj) use v_wmma_f32_16x16x32_f16
//   (wave32 WMMA, f16 operands, f32 accumulate).
// - Weight fragments are staged in LDS (f16) once per block; the WMMA inner
//   loops read A from LDS and gather only activations from global.
// - The selective scan uses a 3-phase chunked linear-recurrence scan
//   (64 chunks x 343 steps) for ~64x more parallelism than a flat scan.
// ---------------------------------------------------------------------------

typedef __attribute__((ext_vector_type(16))) _Float16 v16h;
typedef __attribute__((ext_vector_type(8)))  float    v8f;

#define BQ   4
#define S_   56
#define SD   28
#define LQ   (28*28*28)     // 21952
#define LT16 (LQ/16)        // 1372
#define NCH  64             // scan chunks
#define CLN  343            // chunk length (64*343 == 21952)
#define EPSQ 1e-5f

// ---- WMMA lane/slot maps (ISA 7.12.2, wave32) -----------------------------
// A (16x32 f16): lane L holds row M=L&15; slot s -> K = (L<16?0:8) + (s<8?s:s+8)
// B (32x16 f16): lane L holds col N=L&15; slot s -> K = (L<16?0:8) + (s<8?s:s+8)
// D (16x16 f32): VGPR r -> row M = r + (L<16?0:8), col N = L&15
__device__ __forceinline__ int koffA(int lane, int s) {
  return ((lane & 16) ? 8 : 0) + (s < 8 ? s : s + 8);
}
__device__ __forceinline__ int koffB(int lane, int s) {
  return ((lane & 16) ? 8 : 0) + (s < 8 ? s : s + 8);
}
__device__ __forceinline__ v8f wmma_f16(v16h a, v16h b, v8f c) {
  return __builtin_amdgcn_wmma_f32_16x16x32_f16(false, a, false, b, (short)0, c,
                                                false, false);
}

// ---------------------------------------------------------------------------
// Stage 1: patch-merge + 1x1x1 reduction as a WMMA GEMM.
// grid = (ceil(LT16/8), B*4). blockIdx.y fixes (b, mt); the 16x256 slice of
// red_w is staged once in LDS as f16.
// ---------------------------------------------------------------------------
__global__ void k_pm_gemm(const float* __restrict__ x,
                          const float* __restrict__ red_w,
                          float* __restrict__ xm) {
  int b = blockIdx.y >> 2, mt = blockIdx.y & 3;
  __shared__ _Float16 sA[256 * 16];   // [K][m]
  for (int e = threadIdx.x; e < 256 * 16; e += 256) {
    int m = e >> 8, K = e & 255;
    sA[K * 16 + m] = (_Float16)red_w[(mt * 16 + m) * 256 + K];
  }
  __syncthreads();

  int lane = threadIdx.x & 31, wid = threadIdx.x >> 5;
  int lt = blockIdx.x * 8 + wid;
  if (lt >= LT16) return;
  int n = lane & 15;
  int l = lt * 16 + n;
  int z = l / (SD * SD), y = (l / SD) % SD, xq = l % SD;

  v8f acc = {0.f, 0.f, 0.f, 0.f, 0.f, 0.f, 0.f, 0.f};
  for (int kb = 0; kb < 256; kb += 32) {
    v16h av, bv;
#pragma unroll
    for (int s = 0; s < 16; ++s) {
      av[s] = sA[(kb + koffA(lane, s)) * 16 + n];
      int Kb = kb + koffB(lane, s);
      int p = Kb >> 5, cin = Kb & 31;
      // shifts for parts [(0,0,0),(1,0,0),(0,1,0),(0,0,1),(1,1,0),(1,0,1),(0,1,1),(1,1,1)]
      int di = (178 >> p) & 1, dj = (212 >> p) & 1, dk = (232 >> p) & 1;
      int zz = 2 * z + di, yy = 2 * y + dj, xx = 2 * xq + dk;
      bv[s] = (_Float16)x[(((b * 32 + cin) * S_ + zz) * S_ + yy) * S_ + xx];
    }
    acc = wmma_f16(av, bv, acc);
  }
#pragma unroll
  for (int r = 0; r < 8; ++r) {
    int m = mt * 16 + r + ((lane & 16) ? 8 : 0);
    xm[(b * 64 + m) * LQ + l] = acc[r];
  }
}

// ---------------------------------------------------------------------------
// 3x3x3 SAME conv as implicit GEMM; K = CH*27 ordered k = t*CH + c so a
// 32-wide K-step stays inside one tap. Weight slice (16 x CH*27) staged in
// LDS as f16 per block. grid = (ceil(LT16/8), B*(CH/16)).
// ---------------------------------------------------------------------------
template <int CH>
__global__ void k_conv(const _Float16* __restrict__ inh,
                       const float* __restrict__ w,
                       const float* __restrict__ bias,
                       const float* __restrict__ resid,
                       float* __restrict__ out) {
  const int MT = CH / 16;
  const int K27 = CH * 27;
  int b = blockIdx.y / MT, mt = blockIdx.y % MT;
  __shared__ _Float16 sA[K27 * 16];   // [K][m]
  for (int e = threadIdx.x; e < K27 * 16; e += 256) {
    int m = e / K27, K = e % K27;
    int t = K / CH, c = K % CH;
    sA[K * 16 + m] = (_Float16)w[((mt * 16 + m) * CH + c) * 27 + t];
  }
  __syncthreads();

  int lane = threadIdx.x & 31, wid = threadIdx.x >> 5;
  int lt = blockIdx.x * 8 + wid;
  if (lt >= LT16) return;
  int n = lane & 15;
  int l = lt * 16 + n;
  int z = l / (SD * SD), y = (l / SD) % SD, xq = l % SD;

  v8f acc = {0.f, 0.f, 0.f, 0.f, 0.f, 0.f, 0.f, 0.f};
  for (int kb = 0; kb < K27; kb += 32) {
    int t = kb / CH;                       // tap constant within a step
    int dz = t / 9 - 1, dy = (t % 9) / 3 - 1, dx = t % 3 - 1;
    int zz = z + dz, yy = y + dy, xx = xq + dx;
    bool inb = (unsigned)zz < SD && (unsigned)yy < SD && (unsigned)xx < SD;
    int nb = (zz * SD + yy) * SD + xx;
    v16h av, bv;
#pragma unroll
    for (int s = 0; s < 16; ++s) {
      av[s] = sA[(kb + koffA(lane, s)) * 16 + n];
      int cb2 = kb + koffB(lane, s) - t * CH;
      bv[s] = inb ? inh[(b * CH + cb2) * LQ + nb] : (_Float16)0.f;
    }
    acc = wmma_f16(av, bv, acc);
  }
#pragma unroll
  for (int r = 0; r < 8; ++r) {
    int m = mt * 16 + r + ((lane & 16) ? 8 : 0);
    int idx = (b * CH + m) * LQ + l;
    float v = acc[r] + bias[m];
    if (resid) v += resid[idx];
    out[idx] = v;
  }
}

// ---------------------------------------------------------------------------
// Generic channel GEMM: out[b,m,l] = act( sum_k W[m,k]*inh[b,k,l] + bias[m] )
// grid = (ceil(LT16/8), B*mtiles); the block's 16xKc weight slice is staged
// in LDS (rows beyond M zero-padded). Kc is 32 for every use here.
// ---------------------------------------------------------------------------
__global__ void k_cgemm(const _Float16* __restrict__ inh,
                        const float* __restrict__ W,
                        const float* __restrict__ bias,
                        float* __restrict__ out, int M, int Kc, int act) {
  int mtiles = (M + 15) >> 4;
  int b = blockIdx.y / mtiles, mt = blockIdx.y % mtiles;
  __shared__ _Float16 sW[16 * 32];    // [m][K], Kc <= 32
  for (int e = threadIdx.x; e < 16 * Kc; e += 256) {
    int m = e / Kc, K = e % Kc;
    int row = mt * 16 + m;
    sW[m * Kc + K] = (row < M) ? (_Float16)W[row * Kc + K] : (_Float16)0.f;
  }
  __syncthreads();

  int lane = threadIdx.x & 31, wid = threadIdx.x >> 5;
  int lt = blockIdx.x * 8 + wid;
  if (lt >= LT16) return;
  int n = lane & 15;
  int l = lt * 16 + n;

  v8f acc = {0.f, 0.f, 0.f, 0.f, 0.f, 0.f, 0.f, 0.f};
  for (int kb = 0; kb < Kc; kb += 32) {
    v16h av, bv;
#pragma unroll
    for (int s = 0; s < 16; ++s) {
      av[s] = sW[n * Kc + kb + koffA(lane, s)];
      bv[s] = inh[(b * Kc + kb + koffB(lane, s)) * LQ + l];
    }
    acc = wmma_f16(av, bv, acc);
  }
#pragma unroll
  for (int r = 0; r < 8; ++r) {
    int m = mt * 16 + r + ((lane & 16) ? 8 : 0);
    if (m < M) {
      float v = acc[r] + (bias ? bias[m] : 0.f);
      if (act == 1) v = fmaxf(v, 0.f);
      out[(b * M + m) * LQ + l] = v;
    }
  }
}

// ---------------------------------------------------------------------------
// GroupNorm statistics (biased variance, matches jnp.var).
// ---------------------------------------------------------------------------
__global__ void k_gn_stats(const float* __restrict__ in, float* __restrict__ stats,
                           int CH, int G) {
  int bg = blockIdx.x;
  int b = bg / G, g = bg % G;
  int cpg = CH / G;
  int cnt = cpg * LQ;
  float s = 0.f, q = 0.f;
  for (int e = threadIdx.x; e < cnt; e += blockDim.x) {
    int c = g * cpg + e / LQ;
    float v = in[(b * CH + c) * LQ + (e % LQ)];
    s += v; q += v * v;
  }
  __shared__ float ss[256], sq[256];
  ss[threadIdx.x] = s; sq[threadIdx.x] = q;
  __syncthreads();
  for (int o = 128; o > 0; o >>= 1) {
    if ((int)threadIdx.x < o) { ss[threadIdx.x] += ss[threadIdx.x + o]; sq[threadIdx.x] += sq[threadIdx.x + o]; }
    __syncthreads();
  }
  if (threadIdx.x == 0) {
    float m = ss[0] / cnt, v = sq[0] / cnt - m * m;
    stats[bg * 2] = m;
    stats[bg * 2 + 1] = rsqrtf(v + EPSQ);
  }
}

// GroupNorm apply + ReLU; writes f32 and/or f16 (either may be null).
__global__ void k_gn_apply(const float* __restrict__ in, const float* __restrict__ stats,
                           const float* __restrict__ gam, const float* __restrict__ bet,
                           float* __restrict__ outf, _Float16* __restrict__ outh,
                           int CH, int G) {
  int idx = blockIdx.x * blockDim.x + threadIdx.x;
  int N = BQ * CH * LQ;
  if (idx >= N) return;
  int b = idx / (CH * LQ);
  int c = (idx / LQ) % CH;
  int g = c / (CH / G);
  float m = stats[(b * G + g) * 2], r = stats[(b * G + g) * 2 + 1];
  float v = (in[idx] - m) * r * gam[c] + bet[c];
  v = fmaxf(v, 0.f);
  if (outf) outf[idx] = v;
  if (outh) outh[idx] = (_Float16)v;
}

// Eval-mode BatchNorm (affine) + optional ReLU -> f16.
__global__ void k_bn_apply(const float* __restrict__ in, int inCH, int chOff,
                           const float* __restrict__ g, const float* __restrict__ bta,
                           const float* __restrict__ mn, const float* __restrict__ vr,
                           _Float16* __restrict__ outh, int relu) {
  int idx = blockIdx.x * blockDim.x + threadIdx.x;
  int N = BQ * 32 * LQ;
  if (idx >= N) return;
  int b = idx / (32 * LQ);
  int c = (idx / LQ) % 32;
  int l = idx % LQ;
  float sc = g[c] * rsqrtf(vr[c] + EPSQ);
  float v = in[(b * inCH + chOff + c) * LQ + l] * sc + (bta[c] - mn[c] * sc);
  if (relu) v = fmaxf(v, 0.f);
  outh[idx] = (_Float16)v;
}

// LayerNorm over the 32 right-branch channels of xr -> f16 (B,32,L).
__global__ void k_ln_right(const float* __restrict__ xr, const float* __restrict__ g,
                           const float* __restrict__ bb, _Float16* __restrict__ outh) {
  int idx = blockIdx.x * blockDim.x + threadIdx.x;
  int N = BQ * LQ;
  if (idx >= N) return;
  int b = idx / LQ, l = idx % LQ;
  float v[32], s = 0.f, q = 0.f;
#pragma unroll
  for (int c = 0; c < 32; ++c) {
    float t = xr[(b * 64 + 32 + c) * LQ + l];
    v[c] = t; s += t; q += t * t;
  }
  float m = s / 32.f, r = rsqrtf(q / 32.f - m * m + EPSQ);
#pragma unroll
  for (int c = 0; c < 32; ++c)
    outh[(b * 32 + c) * LQ + l] = (_Float16)((v[c] - m) * r * g[c] + bb[c]);
}

// Depthwise 3x3x3 conv (groups=DI) + SiLU, from xc = xz[:, :32].
__global__ void k_dwconv_silu(const float* __restrict__ xz, const float* __restrict__ w,
                              const float* __restrict__ cb, float* __restrict__ xf,
                              _Float16* __restrict__ xfh) {
  int idx = blockIdx.x * blockDim.x + threadIdx.x;
  int N = BQ * 32 * LQ;
  if (idx >= N) return;
  int b = idx / (32 * LQ), c = (idx / LQ) % 32, l = idx % LQ;
  int z = l / (SD * SD), y = (l / SD) % SD, xq = l % SD;
  float acc = cb[c];
#pragma unroll
  for (int t = 0; t < 27; ++t) {
    int zz = z + t / 9 - 1, yy = y + (t % 9) / 3 - 1, xx = xq + t % 3 - 1;
    if ((unsigned)zz < SD && (unsigned)yy < SD && (unsigned)xx < SD)
      acc += w[c * 27 + t] * xz[(b * 64 + c) * LQ + (zz * SD + yy) * SD + xx];
  }
  float s = acc / (1.f + expf(-acc));
  xf[idx] = s;
  xfh[idx] = (_Float16)s;
}

// delta[b,k,d,l] = softplus(dt_w[k,d,0]*dts0 + dt_w[k,d,1]*dts1 + dt_b[k,d])
__global__ void k_delta(const float* __restrict__ xdbl, const float* __restrict__ dw,
                        const float* __restrict__ db, float* __restrict__ delta) {
  int idx = blockIdx.x * blockDim.x + threadIdx.x;
  int N = BQ * 2 * 32 * LQ;
  if (idx >= N) return;
  int b = idx / (2 * 32 * LQ);
  int k = (idx / (32 * LQ)) % 2;
  int d = (idx / LQ) % 32;
  int l = idx % LQ;
  float d0 = xdbl[(b * 68 + k * 34 + 0) * LQ + l];
  float d1 = xdbl[(b * 68 + k * 34 + 1) * LQ + l];
  float v = dw[(k * 32 + d) * 2] * d0 + dw[(k * 32 + d) * 2 + 1] * d1 + db[k * 32 + d];
  delta[idx] = (v > 20.f) ? v : log1pf(expf(v));
}

// Repack B/C to ((b*2+k)*L + t)*16 + n so 16 scan lanes read 64 contiguous B.
__global__ void k_repack(const float* __restrict__ xdbl, float* __restrict__ Bt,
                         float* __restrict__ Ct) {
  int idx = blockIdx.x * blockDim.x + threadIdx.x;
  int N = BQ * 2 * LQ * 16;
  if (idx >= N) return;
  int n = idx & 15;
  int t = (idx >> 4) % LQ;
  int k = (idx / (16 * LQ)) % 2;
  int b = idx / (2 * 16 * LQ);
  Bt[idx] = xdbl[(b * 68 + k * 34 + 2 + n) * LQ + t];
  Ct[idx] = xdbl[(b * 68 + k * 34 + 18 + n) * LQ + t];
}

// ---------------------------------------------------------------------------
// Chunked selective scan. Per (g,d,n) the recurrence h_t = a_t*h + b_t is
// scalar-linear, so: phase 1 scans each of 64 chunks from h0=0 recording the
// chunk transfer P = prod(a); phase 2 composes chunk states sequentially
// (4096 scalars x 64, trivial); phase 3 re-scans chunks from their true h0
// and emits y via a half-wave (16 states) reduction. k=1 groups traverse the
// natural-order arrays with reversed time index (reversal-free).
// ---------------------------------------------------------------------------
__global__ void k_scan1(const float* __restrict__ delta, const float* __restrict__ xf,
                        const float* __restrict__ Bt, const float* __restrict__ A_log,
                        float* __restrict__ ap, float* __restrict__ hp) {
  int idx = blockIdx.x * blockDim.x + threadIdx.x;  // 262144 = 256*64*16
  int n = idx & 15;
  int j = (idx >> 4) & (NCH - 1);
  int pair = idx >> 10;                             // g*32 + d
  int d = pair & 31, g = pair >> 5;
  int k = g & 1, b = g >> 1;
  float A = -expf(A_log[(k * 32 + d) * 16 + n]);
  const float* dptr = delta + (size_t)pair * LQ;
  const float* uptr = xf + ((size_t)(b * 32 + d)) * LQ;
  const float* bptr = Bt + (size_t)g * LQ * 16;
  float h = 0.f, P = 1.f;
  for (int ii = 0; ii < CLN; ++ii) {
    int i = j * CLN + ii;
    int t = k ? (LQ - 1 - i) : i;
    float dv = dptr[t], u = uptr[t];
    float Bv = bptr[t * 16 + n];
    float a = expf(dv * A);
    h = a * h + dv * u * Bv;
    P *= a;
  }
  ap[((size_t)pair * 16 + n) * NCH + j] = P;
  hp[((size_t)pair * 16 + n) * NCH + j] = h;
}

// Phase 2: sequential compose over chunks; overwrites ap[j] with the chunk's
// true initial state h0.
__global__ void k_scan2(float* __restrict__ ap, const float* __restrict__ hp) {
  int s = blockIdx.x * blockDim.x + threadIdx.x;    // 4096 = 256*16
  float h = 0.f;
  for (int j = 0; j < NCH; ++j) {
    float a = ap[(size_t)s * NCH + j];
    float bb = hp[(size_t)s * NCH + j];
    ap[(size_t)s * NCH + j] = h;
    h = a * h + bb;
  }
}

__global__ void k_scan3(const float* __restrict__ delta, const float* __restrict__ xf,
                        const float* __restrict__ Bt, const float* __restrict__ Ct,
                        const float* __restrict__ A_log, const float* __restrict__ Ds,
                        const float* __restrict__ ap, float* __restrict__ ys) {
  int idx = blockIdx.x * blockDim.x + threadIdx.x;  // 262144
  int n = idx & 15;
  int j = (idx >> 4) & (NCH - 1);
  int pair = idx >> 10;
  int d = pair & 31, g = pair >> 5;
  int k = g & 1, b = g >> 1;
  float A = -expf(A_log[(k * 32 + d) * 16 + n]);
  float Dv = Ds[k * 32 + d];
  const float* dptr = delta + (size_t)pair * LQ;
  const float* uptr = xf + ((size_t)(b * 32 + d)) * LQ;
  const float* bptr = Bt + (size_t)g * LQ * 16;
  const float* cptr = Ct + (size_t)g * LQ * 16;
  float* yptr = ys + (size_t)pair * LQ;
  float h = ap[((size_t)pair * 16 + n) * NCH + j];
  int st = k ? -1 : 1;
  for (int ii = 0; ii < CLN; ++ii) {
    int i = j * CLN + ii;
    int t = k ? (LQ - 1 - i) : i;
    if (ii + 64 < CLN) {
      __builtin_prefetch(&bptr[(t + 64 * st) * 16 + n], 0, 1);
      __builtin_prefetch(&cptr[(t + 64 * st) * 16 + n], 0, 1);
    }
    float dv = dptr[t], u = uptr[t];
    float Bv = bptr[t * 16 + n];
    float Cv = cptr[t * 16 + n];
    h = expf(dv * A) * h + dv * u * Bv;
    float p = h * Cv;
    p += __shfl_xor(p, 1, 16);
    p += __shfl_xor(p, 2, 16);
    p += __shfl_xor(p, 4, 16);
    p += __shfl_xor(p, 8, 16);
    if (n == 0) yptr[t] = p + Dv * u;
  }
}

// ysum = fwd + bwd, LN over 32 ch, gate with silu(z) -> f16 for out_proj.
__global__ void k_gate_ln(const float* __restrict__ ys, const float* __restrict__ xz,
                          const float* __restrict__ g, const float* __restrict__ bb,
                          _Float16* __restrict__ gatedh) {
  int idx = blockIdx.x * blockDim.x + threadIdx.x;
  int N = BQ * LQ;
  if (idx >= N) return;
  int b = idx / LQ, l = idx % LQ;
  float v[32], s = 0.f, q = 0.f;
#pragma unroll
  for (int d = 0; d < 32; ++d) {
    float t = ys[((size_t)(b * 2 + 0) * 32 + d) * LQ + l] +
              ys[((size_t)(b * 2 + 1) * 32 + d) * LQ + l];
    v[d] = t; s += t; q += t * t;
  }
  float m = s / 32.f, r = rsqrtf(q / 32.f - m * m + EPSQ);
#pragma unroll
  for (int d = 0; d < 32; ++d) {
    float zn = xz[(b * 64 + 32 + d) * LQ + l];
    float gate = zn / (1.f + expf(-zn));
    gatedh[(b * 32 + d) * LQ + l] = (_Float16)(((v[d] - m) * r * g[d] + bb[d]) * gate);
  }
}

// concat(left,right) -> channel shuffle (groups=2) -> + xr residual.
__global__ void k_final(const float* __restrict__ lbuf, const float* __restrict__ rbuf,
                        const float* __restrict__ xr, float* __restrict__ out) {
  int idx = blockIdx.x * blockDim.x + threadIdx.x;
  int N = BQ * 64 * LQ;
  if (idx >= N) return;
  int b = idx / (64 * LQ);
  int c = (idx / LQ) % 64;
  int l = idx % LQ;
  int c2 = c >> 1;
  float src = (c & 1) ? rbuf[(b * 32 + c2) * LQ + l] : lbuf[(b * 32 + c2) * LQ + l];
  out[idx] = src + xr[idx];
}

// ---------------------------------------------------------------------------
extern "C" void kernel_launch(void* const* d_in, const int* in_sizes, int n_in,
                              void* d_out, int out_size, void* d_ws, size_t ws_size,
                              hipStream_t stream) {
  (void)in_sizes; (void)n_in; (void)out_size; (void)ws_size;
  const float* x        = (const float*)d_in[0];
  const float* red_w    = (const float*)d_in[1];
  const float* gn1_g    = (const float*)d_in[2];
  const float* gn1_b    = (const float*)d_in[3];
  const float* c1_w     = (const float*)d_in[4];
  const float* c1_b     = (const float*)d_in[5];
  const float* gn2_g    = (const float*)d_in[6];
  const float* gn2_b    = (const float*)d_in[7];
  const float* c2_w     = (const float*)d_in[8];
  const float* c2_b     = (const float*)d_in[9];
  const float* gn3_g    = (const float*)d_in[10];
  const float* gn3_b    = (const float*)d_in[11];
  const float* ln1_g    = (const float*)d_in[12];
  const float* ln1_b    = (const float*)d_in[13];
  const float* bnA_g    = (const float*)d_in[14];
  const float* bnA_b    = (const float*)d_in[15];
  const float* bnA_m    = (const float*)d_in[16];
  const float* bnA_v    = (const float*)d_in[17];
  const float* cA_w     = (const float*)d_in[18];
  const float* cA_b     = (const float*)d_in[19];
  const float* bnB_g    = (const float*)d_in[20];
  const float* bnB_b    = (const float*)d_in[21];
  const float* bnB_m    = (const float*)d_in[22];
  const float* bnB_v    = (const float*)d_in[23];
  const float* cB_w     = (const float*)d_in[24];
  const float* cB_b     = (const float*)d_in[25];
  const float* bnC_g    = (const float*)d_in[26];
  const float* bnC_b    = (const float*)d_in[27];
  const float* bnC_m    = (const float*)d_in[28];
  const float* bnC_v    = (const float*)d_in[29];
  const float* cC_w     = (const float*)d_in[30];
  const float* cC_b     = (const float*)d_in[31];
  const float* in_pw    = (const float*)d_in[32];
  const float* in_pb    = (const float*)d_in[33];
  const float* conv_w   = (const float*)d_in[34];
  const float* conv_b   = (const float*)d_in[35];
  const float* x_proj_w = (const float*)d_in[36];  // (2,34,32) == flat (68,32)
  const float* dt_pw    = (const float*)d_in[37];
  const float* dt_pb    = (const float*)d_in[38];
  const float* A_log    = (const float*)d_in[39];
  const float* Ds       = (const float*)d_in[40];
  const float* on_g     = (const float*)d_in[41];
  const float* on_b     = (const float*)d_in[42];
  const float* out_pw   = (const float*)d_in[43];
  const float* out_pb   = (const float*)d_in[44];

  // ---- workspace layout (256B aligned bump allocator) ----
  const size_t F64 = (size_t)BQ * 64 * LQ;  // 5,619,712
  const size_t F32 = (size_t)BQ * 32 * LQ;  // 2,809,856
  const size_t FSC = (size_t)256 * 16 * NCH; // 262,144 scan chunk scalars
  char* W = (char*)d_ws;
  size_t off = 0;
  auto alloc = [&](size_t bytes) -> void* {
    void* p = W + off;
    off += (bytes + 255) & ~(size_t)255;
    return p;
  };
  float*    xm     = (float*)alloc(F64 * 4);            // reused later as xz
  float*    tmpA   = (float*)alloc(F64 * 4);            // conv1 out; reused as ys
  float*    xr     = (float*)alloc(F64 * 4);
  _Float16* h16    = (_Float16*)alloc(F64 * 2);
  float*    stats  = (float*)alloc(256 * 4);
  float*    ltmp   = (float*)alloc(F32 * 4);
  float*    lbuf   = (float*)alloc(F32 * 4);
  _Float16* lh     = (_Float16*)alloc(F32 * 2);
  _Float16* rh     = (_Float16*)alloc(F32 * 2);
  float*    xf     = (float*)alloc(F32 * 4);
  _Float16* xfh    = (_Float16*)alloc(F32 * 2);
  float*    xdbl   = (float*)alloc((size_t)BQ * 68 * LQ * 4);
  float*    delta  = (float*)alloc(F64 * 4);
  float*    Btb    = (float*)alloc(F32 * 4 * 2);        // B then C halves
  float*    Ctb    = Btb + F32;
  float*    apb    = (float*)alloc(FSC * 4);
  float*    hpb    = (float*)alloc(FSC * 4);
  _Float16* gatedh = (_Float16*)alloc(F32 * 2);
  float*    rbuf   = (float*)alloc(F32 * 4);
  float*    xz     = xm;    // alias: xm dead after gn3
  float*    ys     = tmpA;  // alias: tmpA dead after gn2-apply

  dim3 blk(256);
  dim3 gT64(172, BQ * 4);               // 64-ch tiles: (lt/8, b*mt)
  dim3 gT32(172, BQ * 2);
  dim3 gT68(172, BQ * 5);               // M=68 -> 5 m-tiles
  int eb64 = (int)((F64 + 255) / 256);
  int eb32 = (int)((F32 + 255) / 256);
  int ebBL = (BQ * LQ + 255) / 256;
  int ebRP = (BQ * 2 * 16 * LQ + 255) / 256;

  // ---- Stage 1: patch merge + reduction, ResConvBlock, gn3 ----
  k_pm_gemm<<<gT64, blk, 0, stream>>>(x, red_w, xm);
  k_gn_stats<<<32, blk, 0, stream>>>(xm, stats, 64, 8);
  k_gn_apply<<<eb64, blk, 0, stream>>>(xm, stats, gn1_g, gn1_b, nullptr, h16, 64, 8);
  k_conv<64><<<gT64, blk, 0, stream>>>(h16, c1_w, c1_b, nullptr, tmpA);
  k_gn_stats<<<32, blk, 0, stream>>>(tmpA, stats, 64, 8);
  k_gn_apply<<<eb64, blk, 0, stream>>>(tmpA, stats, gn2_g, gn2_b, nullptr, h16, 64, 8);
  k_conv<64><<<gT64, blk, 0, stream>>>(h16, c2_w, c2_b, xm, xm);  // += xm (in-place ok)
  k_gn_stats<<<32, blk, 0, stream>>>(xm, stats, 64, 8);
  k_gn_apply<<<eb64, blk, 0, stream>>>(xm, stats, gn3_g, gn3_b, xr, nullptr, 64, 8);

  // ---- Left branch: bnA -> convA -> relu(bnB) -> convB -> relu(bnC) -> cC ----
  k_bn_apply<<<eb32, blk, 0, stream>>>(xr, 64, 0, bnA_g, bnA_b, bnA_m, bnA_v, lh, 0);
  k_conv<32><<<gT32, blk, 0, stream>>>(lh, cA_w, cA_b, nullptr, ltmp);
  k_bn_apply<<<eb32, blk, 0, stream>>>(ltmp, 32, 0, bnB_g, bnB_b, bnB_m, bnB_v, lh, 1);
  k_conv<32><<<gT32, blk, 0, stream>>>(lh, cB_w, cB_b, nullptr, ltmp);
  k_bn_apply<<<eb32, blk, 0, stream>>>(ltmp, 32, 0, bnC_g, bnC_b, bnC_m, bnC_v, lh, 1);
  k_cgemm<<<gT32, blk, 0, stream>>>(lh, cC_w, cC_b, lbuf, 32, 32, 1);

  // ---- Right branch: SS3D ----
  k_ln_right<<<ebBL, blk, 0, stream>>>(xr, ln1_g, ln1_b, rh);
  k_cgemm<<<gT64, blk, 0, stream>>>(rh, in_pw, in_pb, xz, 64, 32, 0);
  k_dwconv_silu<<<eb32, blk, 0, stream>>>(xz, conv_w, conv_b, xf, xfh);
  k_cgemm<<<gT68, blk, 0, stream>>>(xfh, x_proj_w, nullptr, xdbl, 68, 32, 0);
  k_delta<<<eb64, blk, 0, stream>>>(xdbl, dt_pw, dt_pb, delta);
  k_repack<<<ebRP, blk, 0, stream>>>(xdbl, Btb, Ctb);
  k_scan1<<<1024, blk, 0, stream>>>(delta, xf, Btb, A_log, apb, hpb);
  k_scan2<<<16, blk, 0, stream>>>(apb, hpb);
  k_scan3<<<1024, blk, 0, stream>>>(delta, xf, Btb, Ctb, A_log, Ds, apb, ys);
  k_gate_ln<<<ebBL, blk, 0, stream>>>(ys, xz, on_g, on_b, gatedh);
  k_cgemm<<<gT32, blk, 0, stream>>>(gatedh, out_pw, out_pb, rbuf, 32, 32, 0);

  // ---- concat + channel shuffle + residual ----
  k_final<<<eb64, blk, 0, stream>>>(lbuf, rbuf, xr, (float*)d_out);
}
